// LGAE_Concat_40699110097055
// MI455X (gfx1250) — compile-verified, matrix operand order checked by reading, not run on hardware
//
#include <hip/hip_runtime.h>
#include <hip/hip_bf16.h>

typedef __attribute__((ext_vector_type(16))) __bf16 v16bf;
typedef __attribute__((ext_vector_type(8)))  __bf16 v8bf;
typedef __attribute__((ext_vector_type(8)))  float  v8f;
typedef __attribute__((ext_vector_type(4)))  float  f32x4;

#define WPB     8      // waves per block (wave32 -> 256 threads)
#define KSPLIT  2      // K split of the big GEMM (deterministic fp32 partials)
#define KSTAGE  128    // K elements staged in LDS per stage (4 WMMA K-steps)
#define ROWB    (KSTAGE * 2)          // bytes per staged row (bf16)
#define RSTRIDE (ROWB + 16)           // +16B pad: 68 dwords -> 4-bank shift/row
#define STAGEB  (128 * RSTRIDE)       // 34816 B per stage buffer

// Raw 32-bit LDS offset of a __shared__ object (for async-to-LDS dest addr).
__device__ __forceinline__ unsigned lds_addr32(const void* p)
{
    return (unsigned)(unsigned long long)
        (const __attribute__((address_space(3))) char*)p;
}

// Async-copy one 128-row x 256B slice of BT (rows have stride K elements)
// into padded LDS.  2048 16B chunks spread over 256 threads -> 8 per thread.
// Tracked by ASYNCcnt; caller pairs with s_wait_asynccnt + barrier.
__device__ __forceinline__ void stage_fill(const __bf16* __restrict__ BTk, size_t K,
                                           unsigned ldsbase, int tid)
{
#pragma unroll
    for (int j = 0; j < 8; ++j) {
        const int c   = tid + j * 256;        // 16B chunk id
        const int row = c >> 4;               // 16 chunks per 256B row
        const int ci  = c & 15;
        const __bf16* g = BTk + (size_t)row * K + ci * 8;
        const unsigned l = ldsbase + (unsigned)row * RSTRIDE + (unsigned)ci * 16u;
        asm volatile("global_load_async_to_lds_b128 %0, %1, off"
                     :: "v"(l), "v"((unsigned long long)(size_t)g)
                     : "memory");
    }
}

// -----------------------------------------------------------------------------
// C[M,128] = A[M,K](f32 rm) x B[K,128], B supplied transposed: BT[128,K] bf16.
// One wave owns 16 rows x 128 cols (8 f32 accum tiles).  B flows
// global -> (async DMA) -> LDS (double-buffered, padded) -> ds_load_b128,
// shared by all 8 waves; A (adj) is NT-streamed from HBM, 1 stage (4 K-steps)
// of loads issued as one clause, converted to bf16 at use.
// blockIdx.y picks a K chunk; each chunk writes its own fp32 partial.
// -----------------------------------------------------------------------------
template <bool OUT_TRANSPOSED_BF16>
__global__ void __launch_bounds__(256)
gemm_bf16_wmma(const float* __restrict__ A, const __bf16* __restrict__ BT,
               void* __restrict__ Cout, int M, int K, int kchunk)
{
    __shared__ __align__(16) char smem[2 * STAGEB];

    const int tid  = threadIdx.x;
    const int wave = tid >> 5;
    const int lane = tid & 31;
    const int hl   = lane >> 4;          // half-wave: K sub-block select
    const int nrow = lane & 15;          // A-row / B-col within tile
    const int m0   = (blockIdx.x * WPB + wave) * 16;
    const int kb   = blockIdx.y * kchunk;
    const int nstages = kchunk / KSTAGE;

    v8f c[8] = {};
    const float* arow = A + (size_t)(m0 + nrow) * (size_t)K;
    const unsigned lds0 = lds_addr32(smem);

    stage_fill(BT + kb, (size_t)K, lds0, tid);
    asm volatile("s_wait_asynccnt 0" ::: "memory");
    __syncthreads();

    for (int s = 0; s < nstages; ++s) {
        if (s + 1 < nstages)                       // overlap next B stage DMA
            stage_fill(BT + (kb + (s + 1) * KSTAGE), (size_t)K,
                       lds0 + (unsigned)((s + 1) & 1) * STAGEB, tid);

        // A: issue the whole stage's HBM stream (16 x NT b128) up front
        f32x4 fa[16];
#pragma unroll
        for (int kk = 0; kk < 4; ++kk) {
            const float* ap = arow + (kb + s * KSTAGE + kk * 32 + hl * 8);
            fa[kk * 4 + 0] = __builtin_nontemporal_load((const f32x4*)ap);
            fa[kk * 4 + 1] = __builtin_nontemporal_load((const f32x4*)ap + 1);
            fa[kk * 4 + 2] = __builtin_nontemporal_load((const f32x4*)(ap + 16));
            fa[kk * 4 + 3] = __builtin_nontemporal_load((const f32x4*)(ap + 16) + 1);
        }

        const char* sm = smem + (s & 1) * STAGEB;
#pragma unroll
        for (int kk = 0; kk < 4; ++kk) {
            v16bf a;
            a[0]  = (__bf16)fa[kk*4+0].x; a[1]  = (__bf16)fa[kk*4+0].y;
            a[2]  = (__bf16)fa[kk*4+0].z; a[3]  = (__bf16)fa[kk*4+0].w;
            a[4]  = (__bf16)fa[kk*4+1].x; a[5]  = (__bf16)fa[kk*4+1].y;
            a[6]  = (__bf16)fa[kk*4+1].z; a[7]  = (__bf16)fa[kk*4+1].w;
            a[8]  = (__bf16)fa[kk*4+2].x; a[9]  = (__bf16)fa[kk*4+2].y;
            a[10] = (__bf16)fa[kk*4+2].z; a[11] = (__bf16)fa[kk*4+2].w;
            a[12] = (__bf16)fa[kk*4+3].x; a[13] = (__bf16)fa[kk*4+3].y;
            a[14] = (__bf16)fa[kk*4+3].z; a[15] = (__bf16)fa[kk*4+3].w;

            const int koff = kk * 64 + hl * 16;   // byte offset of K sub-block
#pragma unroll
            for (int nt = 0; nt < 8; ++nt) {
                const char* bp = sm + (nt * 16 + nrow) * RSTRIDE + koff;
                v8bf blo = *(const v8bf*)(bp);         // K = +0..7
                v8bf bhi = *(const v8bf*)(bp + 32);    // K = +16..23
                v16bf b  = __builtin_shufflevector(blo, bhi,
                              0, 1, 2, 3, 4, 5, 6, 7, 8, 9, 10, 11, 12, 13, 14, 15);
                c[nt] = __builtin_amdgcn_wmma_f32_16x16x32_bf16(
                            false, a, false, b, (short)0, c[nt], false, false);
            }
        }

        if (s + 1 < nstages) {
            asm volatile("s_wait_asynccnt 0" ::: "memory");
            __syncthreads();
        }
    }

    // C/D layout: VGPR i -> row (hl*8 + i), col = nt*16 + (lane&15)
    if (OUT_TRANSPOSED_BF16) {
        __bf16* O = (__bf16*)Cout;                  // [128, M]
#pragma unroll
        for (int nt = 0; nt < 8; ++nt)
#pragma unroll
            for (int i = 0; i < 8; ++i) {
                const int row = m0 + hl * 8 + i;
                const int col = nt * 16 + nrow;
                O[(size_t)col * (size_t)M + row] = (__bf16)c[nt][i];
            }
    } else {
        float* O = (float*)Cout + (size_t)blockIdx.y * (size_t)M * 128;  // partial y
#pragma unroll
        for (int nt = 0; nt < 8; ++nt)
#pragma unroll
            for (int i = 0; i < 8; ++i) {
                const int row = m0 + hl * 8 + i;
                const int col = nt * 16 + nrow;
                O[(size_t)row * 128 + col] = c[nt][i];
            }
    }
}

// WT[n][k] = bf16(W[k][n]) : W is [K, Nc] f32 row-major
__global__ void transpose_to_bf16(const float* __restrict__ W, __bf16* __restrict__ WT,
                                  int K, int Nc)
{
    int idx = blockIdx.x * blockDim.x + threadIdx.x;
    if (idx >= K * Nc) return;
    int k = idx / Nc, n = idx % Nc;
    WT[(size_t)n * K + k] = (__bf16)W[idx];
}

// v[j] = sum_l W2[j][l] * W3[l]   (j in [0,256), l in [0,128))
__global__ void fuse_w2w3(const float* __restrict__ W2, const float* __restrict__ W3,
                          float* __restrict__ v)
{
    int j = blockIdx.x * blockDim.x + threadIdx.x;
    if (j >= 256) return;
    float acc = 0.f;
#pragma unroll 4
    for (int l = 0; l < 128; ++l) acc += W2[j * 128 + l] * W3[l];
    v[j] = acc;
}

// one wave32 per node: sum the two K-split partials of Z, then
// s[i]=dot(relu(Z[i]),v[0:128]), t[i]=dot(relu(Z[i]),v[128:256])
__global__ void __launch_bounds__(256)
node_scores(const float* __restrict__ Z0, const float* __restrict__ Z1,
            const float* __restrict__ v,
            float* __restrict__ s, float* __restrict__ t)
{
    const int wave = threadIdx.x >> 5;
    const int lane = threadIdx.x & 31;
    const int node = blockIdx.x * WPB + wave;

    f32x4 za = *(const f32x4*)(Z0 + (size_t)node * 128 + lane * 4);
    f32x4 zb = *(const f32x4*)(Z1 + (size_t)node * 128 + lane * 4);
    f32x4 vt = *(const f32x4*)(v + lane * 4);
    f32x4 vb = *(const f32x4*)(v + 128 + lane * 4);
    f32x4 r;
    r.x = fmaxf(za.x + zb.x, 0.f); r.y = fmaxf(za.y + zb.y, 0.f);
    r.z = fmaxf(za.z + zb.z, 0.f); r.w = fmaxf(za.w + zb.w, 0.f);
    float ps = r.x * vt.x + r.y * vt.y + r.z * vt.z + r.w * vt.w;
    float pt = r.x * vb.x + r.y * vb.y + r.z * vb.z + r.w * vb.w;
#pragma unroll
    for (int off = 16; off > 0; off >>= 1) {
        ps += __shfl_xor(ps, off, 32);
        pt += __shfl_xor(pt, off, 32);
    }
    if (lane == 0) { s[node] = ps; t[node] = pt; }
}

// out[e] = sigmoid(s[e0] + t[e1]); edges = concat(pos, neg)
__global__ void edge_scores(const int* __restrict__ e_pos, const int* __restrict__ e_neg,
                            const float* __restrict__ s, const float* __restrict__ t,
                            float* __restrict__ out, int Epos, int Etot)
{
    int e = blockIdx.x * blockDim.x + threadIdx.x;
    if (e >= Etot) return;
    const int* ep = (e < Epos) ? (e_pos + 2 * (size_t)e) : (e_neg + 2 * (size_t)(e - Epos));
    float h = s[ep[0]] + t[ep[1]];
    out[e] = 1.f / (1.f + __expf(-h));
}

extern "C" void kernel_launch(void* const* d_in, const int* in_sizes, int n_in,
                              void* d_out, int out_size, void* d_ws, size_t ws_size,
                              hipStream_t stream)
{
    const float* X   = (const float*)d_in[0];   // [16384, 512]
    const float* adj = (const float*)d_in[1];   // [16384, 16384]
    const float* W1  = (const float*)d_in[2];   // [512, 128]
    const float* W2  = (const float*)d_in[3];   // [256, 128]
    const float* W3  = (const float*)d_in[4];   // [128, 1]
    const int*  e_pos = (const int*)d_in[5];    // [Epos, 2]
    const int*  e_neg = (const int*)d_in[6];    // [Eneg, 2]
    float* out = (float*)d_out;

    const int N = 16384, K1 = 512, H = 128;
    const int Epos = in_sizes[5] / 2;
    const int Eneg = in_sizes[6] / 2;
    const int Etot = Epos + Eneg;

    // workspace carve-up (all 16B-aligned offsets), ~20.5 MB total
    char* ws = (char*)d_ws;
    __bf16* XWT = (__bf16*)ws; ws += (size_t)H * N * sizeof(__bf16);            // 4 MB
    __bf16* WT  = (__bf16*)ws; ws += (size_t)H * K1 * sizeof(__bf16);           // 128 KB
    float*  Zp  = (float*)ws;  ws += (size_t)KSPLIT * N * H * sizeof(float);    // 16 MB
    float*  v   = (float*)ws;  ws += 256 * sizeof(float);
    float*  sb  = (float*)ws;  ws += (size_t)N * sizeof(float);
    float*  tb  = (float*)ws;

    // 1) WT = bf16(W1^T)   [128, 512]
    transpose_to_bf16<<<(K1 * H + 255) / 256, 256, 0, stream>>>(W1, WT, K1, H);
    // 2) XWT = bf16((X @ W1)^T)   [128, 16384]   (WMMA, 4 LDS stages)
    gemm_bf16_wmma<true><<<dim3(N / (16 * WPB), 1), 256, 0, stream>>>(X, WT, XWT, N, K1, K1);
    // 3) Zp[y] = adj @ XW over K chunk y  (WMMA; adj streamed once at NT,
    //    B async-DMA'd to LDS, 2048 waves vs the 23.3 TB/s roofline)
    gemm_bf16_wmma<false><<<dim3(N / (16 * WPB), KSPLIT), 256, 0, stream>>>(adj, XWT, Zp, N, N, N / KSPLIT);
    // 4) v = W2 @ W3   [256]
    fuse_w2w3<<<1, 256, 0, stream>>>(W2, W3, v);
    // 5) per-node relu-dot scores (sums the KSPLIT partials)
    node_scores<<<N / WPB, 256, 0, stream>>>(Zp, Zp + (size_t)N * H, v, sb, tb);
    // 6) per-edge sigmoid(s[e0] + t[e1])
    edge_scores<<<(Etot + 255) / 256, 256, 0, stream>>>(e_pos, e_neg, sb, tb, out, Epos, Etot);
}
